// GCN_24824910971032
// MI455X (gfx1250) — matrix operand order repaired and checked by reading, not compiled, not run on hardware
//
#include <hip/hip_runtime.h>
#include <cstddef>

typedef __attribute__((ext_vector_type(16))) _Float16 v16h;
typedef __attribute__((ext_vector_type(8)))  _Float16 v8h;
typedef __attribute__((ext_vector_type(8)))  float    v8f;

#define DF 128

// ---------------- norm precompute ----------------

__global__ void deg_init(float* deg, int n) {
    int i = blockIdx.x * blockDim.x + threadIdx.x;
    if (i < n) deg[i] = 1.0f;                       // +1 self-loop weight
}

__global__ void deg_accum(const int* __restrict__ dst, const float* __restrict__ ea,
                          float* deg, int E) {
    int e = blockIdx.x * blockDim.x + threadIdx.x;
    if (e < E) atomicAdd(&deg[dst[e]], ea[e]);
}

__global__ void dinv_finish(float* dinv, float* self_norm, int n) {
    int i = blockIdx.x * blockDim.x + threadIdx.x;
    if (i < n) {
        float d = rsqrtf(dinv[i]);                  // deg >= 1, always valid
        dinv[i] = d;
        self_norm[i] = d * d;
    }
}

__global__ void norm_kernel(const int* __restrict__ src, const int* __restrict__ dst,
                            const float* __restrict__ ea, const float* __restrict__ dinv,
                            float* norm, int E) {
    int e = blockIdx.x * blockDim.x + threadIdx.x;
    if (e < E) norm[e] = dinv[src[e]] * ea[e] * dinv[dst[e]];
}

__global__ void zero_f32(float* p, size_t n) {
    size_t i = (size_t)blockIdx.x * blockDim.x + threadIdx.x;
    if (i < n) p[i] = 0.0f;
}

// ---------------- WMMA GEMM:  out[N x ncols] = A[N x 128] * W[128 x ldw] ----------------
// Block = 256 threads = 8 waves, owns 64 A-rows. A staged fp32->f16 in LDS (row-padded,
// bank-conflict-free frag reads). W staged in LDS *pre-swizzled into B-fragment order*
// so the MAC loop touches global memory zero times. Wave w owns col tile [16w,16w+16),
// iterates 4 row tiles, reusing each B fragment across 4 v_wmma_f32_16x16x32_f16.

#define SA_LDA 136   // 128 + 8 halves pad: row m -> banks 4m..4m+3 for frag reads

__global__ __launch_bounds__(256)
void gemm_wmma(const float* __restrict__ A, const float* __restrict__ W,
               float* __restrict__ out, int nrows, int ldw, int ncols, int ldo) {
    __shared__ __attribute__((aligned(32))) _Float16 sA[64][SA_LDA];
    __shared__ __attribute__((aligned(32))) _Float16 sW[DF * DF];   // frag-ordered; DF*ncols used

    const int tid = threadIdx.x;
    const int m0  = blockIdx.x * 64;

    // ---- stage A: 64 rows x 128 cols; thread -> 8 contiguous floats per row-slab ----
    {
        int rloc = tid >> 4;            // 0..15
        int col0 = (tid & 15) << 3;     // 0,8,...,120
#pragma unroll
        for (int it = 0; it < 4; ++it) {
            int row = rloc + it * 16;
            int r = m0 + row;
            if (r >= nrows) r = nrows - 1;   // clamp tail (stores are guarded below)
            const float4* p = (const float4*)(A + (size_t)r * DF + col0);
            float4 v0 = p[0];
            float4 v1 = p[1];
            sA[row][col0 + 0] = (_Float16)v0.x;
            sA[row][col0 + 1] = (_Float16)v0.y;
            sA[row][col0 + 2] = (_Float16)v0.z;
            sA[row][col0 + 3] = (_Float16)v0.w;
            sA[row][col0 + 4] = (_Float16)v1.x;
            sA[row][col0 + 5] = (_Float16)v1.y;
            sA[row][col0 + 6] = (_Float16)v1.z;
            sA[row][col0 + 7] = (_Float16)v1.w;
        }
    }

    // ---- stage W in B-fragment order (coalesced global reads, scattered LDS writes) ----
    // frag element i of (col-tile ct, k-step s, lane l) holds W[k][n] with
    //   n = ct*16 + (l&15),  k = s*32 + ((l>>4)&1)*16 + i
    {
        int total = DF * ncols;
        for (int j = tid; j < total; j += 256) {
            int k = j / ncols;
            int n = j % ncols;
            int off = ((((n >> 4) * 4 + (k >> 5)) * 32 + (((k >> 4) & 1) * 16) + (n & 15)) << 4)
                      + (k & 15);
            sW[off] = (_Float16)W[(size_t)k * ldw + n];
        }
    }
    __syncthreads();

    const int wave = tid >> 5;
    const int lane = tid & 31;
    const int n0   = wave * 16;
    if (n0 >= ncols) return;            // wave-uniform; idle waves only for layer-2 (ncols=16)

    const int mrow = lane & 15;         // A row within 16-row tile
    const int hi   = lane >> 4;         // half-wave select
    const int ka   = hi * 16;           // A K-base offset (contiguous K order, matches B)
    const int ncol = n0 + (lane & 15);  // output / B column

    v8f c[4];
#pragma unroll
    for (int mt = 0; mt < 4; ++mt)
#pragma unroll
        for (int i = 0; i < 8; ++i) c[mt][i] = 0.0f;

#pragma unroll
    for (int s = 0; s < 4; ++s) {
        const int k0 = s * 32;
        v16h b = *(const v16h*)&sW[((wave * 4 + s) * 32 + lane) << 4];   // 32B aligned
#pragma unroll
        for (int mt = 0; mt < 4; ++mt) {
            const _Float16* arow = &sA[mt * 16 + mrow][0];
            v8h lo = *(const v8h*)(arow + k0 + ka);        // K = k0+ka .. +7
            v8h hp = *(const v8h*)(arow + k0 + ka + 8);    // K = k0+ka+8 .. +15
            v16h a;
#pragma unroll
            for (int i = 0; i < 8; ++i) { a[i] = lo[i]; a[i + 8] = hp[i]; }
            c[mt] = __builtin_amdgcn_wmma_f32_16x16x32_f16(
                        /*neg_a=*/false, a, /*neg_b=*/false, b,
                        /*c_mod=*/(short)0, c[mt], /*reuse_a=*/false, /*reuse_b=*/false);
        }
    }

    // C/D layout: VGPR r -> row r (lanes 0-15) / row r+8 (lanes 16-31), col = lane&15.
#pragma unroll
    for (int mt = 0; mt < 4; ++mt) {
#pragma unroll
        for (int r = 0; r < 8; ++r) {
            int m = m0 + mt * 16 + r + hi * 8;
            if (m < nrows) out[(size_t)m * ldo + ncol] = c[mt][r];
        }
    }
}

// ---------------- edge aggregation ----------------

// One block per edge; 128 lanes = 128 features. Gather t[src] (coalesced, L2-resident),
// scatter-add into agg[dst] with global_atomic_add_f32.
__global__ __launch_bounds__(128)
void scatter128(const float* __restrict__ t, const int* __restrict__ src,
                const int* __restrict__ dst, const float* __restrict__ norm,
                float* agg, int E) {
    int f = threadIdx.x;
    for (int e = blockIdx.x; e < E; e += gridDim.x) {
        int   s = src[e];
        int   d = dst[e];
        float w = norm[e];
        atomicAdd(&agg[(size_t)d * DF + f], t[(size_t)s * DF + f] * w);
    }
}

// 2-feature variant for the output layer (t2 has row stride 16 from padded GEMM).
__global__ __launch_bounds__(128)
void scatter2(const float* __restrict__ t2, const int* __restrict__ src,
              const int* __restrict__ dst, const float* __restrict__ norm,
              float* agg2, int E) {
    int e = blockIdx.x * 64 + (threadIdx.x >> 1);
    int f = threadIdx.x & 1;
    if (e < E) {
        int s = src[e], d = dst[e];
        atomicAdd(&agg2[(size_t)d * 2 + f], t2[(size_t)s * 16 + f] * norm[e]);
    }
}

// ---------------- epilogue ----------------

__global__ void finalize128(const float* __restrict__ agg, const float* __restrict__ t,
                            const float* __restrict__ sn, const float* __restrict__ bias,
                            float* out, int n, int do_relu) {
    size_t i = (size_t)blockIdx.x * blockDim.x + threadIdx.x;
    if (i < (size_t)n * DF) {
        int node = (int)(i >> 7);
        int f    = (int)(i & (DF - 1));
        float v = agg[i] + t[i] * sn[node] + bias[f];
        out[i] = do_relu ? fmaxf(v, 0.0f) : v;
    }
}

__global__ void finalize2(const float* __restrict__ agg2, const float* __restrict__ t2,
                          const float* __restrict__ sn, const float* __restrict__ bias,
                          float* out, int n) {
    int i = blockIdx.x * blockDim.x + threadIdx.x;
    if (i < n * 2) {
        int node = i >> 1;
        int f    = i & 1;
        out[i] = agg2[i] + t2[(size_t)node * 16 + f] * sn[node] + bias[f];
    }
}

__global__ void pad_w2(const float* __restrict__ W2, float* Wp) {
    int i = blockIdx.x * blockDim.x + threadIdx.x;   // 128*16 entries
    if (i < 128 * 16) {
        int r = i >> 4, c = i & 15;
        Wp[i] = (c < 2) ? W2[r * 2 + c] : 0.0f;
    }
}

// ---------------- launcher ----------------

extern "C" void kernel_launch(void* const* d_in, const int* in_sizes, int n_in,
                              void* d_out, int out_size, void* d_ws, size_t ws_size,
                              hipStream_t stream) {
    const float* x  = (const float*)d_in[0];
    const int*   ei = (const int*)d_in[1];
    const float* ea = (const float*)d_in[2];
    const float* W0 = (const float*)d_in[3];
    const float* b0 = (const float*)d_in[4];
    const float* W1 = (const float*)d_in[5];
    const float* b1 = (const float*)d_in[6];
    const float* W2 = (const float*)d_in[7];
    const float* b2 = (const float*)d_in[8];
    float* out = (float*)d_out;

    const int N = in_sizes[0] / DF;     // 50000
    const int E = in_sizes[2];          // 800000
    const int* src = ei;                // edge_index row 0
    const int* dst = ei + E;            // edge_index row 1

    // workspace layout (all offsets stay 16B aligned: N, E multiples of 4)
    float* ws   = (float*)d_ws;
    float* dinv = ws;                      ws += N;
    float* sn   = ws;                      ws += N;
    float* norm = ws;                      ws += E;
    float* t    = ws;                      ws += (size_t)N * DF;
    float* agg  = ws;                      ws += (size_t)N * DF;
    float* h    = ws;                      ws += (size_t)N * DF;
    float* Wp   = ws;                      ws += 128 * 16;
    float* t2   = ws;                      ws += (size_t)N * 16;
    float* agg2 = ws;                      ws += (size_t)N * 2;

    const int TB = 256;
    const int gN    = (N + TB - 1) / TB;
    const int gE    = (E + TB - 1) / TB;
    const int gNF   = (int)(((size_t)N * DF + TB - 1) / TB);
    const int gGemm = (N + 63) / 64;

    // ---- normalization coefficients ----
    deg_init   <<<gN, TB, 0, stream>>>(dinv, N);
    deg_accum  <<<gE, TB, 0, stream>>>(dst, ea, dinv, E);
    dinv_finish<<<gN, TB, 0, stream>>>(dinv, sn, N);
    norm_kernel<<<gE, TB, 0, stream>>>(src, dst, ea, dinv, norm, E);

    // ---- layer 0: h = relu(conv(x, W0, b0)) ----
    zero_f32   <<<gNF, TB, 0, stream>>>(agg, (size_t)N * DF);
    gemm_wmma  <<<gGemm, 256, 0, stream>>>(x, W0, t, N, DF, DF, DF);
    scatter128 <<<E, 128, 0, stream>>>(t, src, dst, norm, agg, E);
    finalize128<<<gNF, TB, 0, stream>>>(agg, t, sn, b0, h, N, 1);

    // ---- layer 1: h = relu(conv(h, W1, b1)) ----
    zero_f32   <<<gNF, TB, 0, stream>>>(agg, (size_t)N * DF);
    gemm_wmma  <<<gGemm, 256, 0, stream>>>(h, W1, t, N, DF, DF, DF);
    scatter128 <<<E, 128, 0, stream>>>(t, src, dst, norm, agg, E);
    finalize128<<<gNF, TB, 0, stream>>>(agg, t, sn, b1, h, N, 1);

    // ---- layer 2: out = conv(h, W2, b2), D_OUT = 2 via 16-col zero-padded W2 ----
    pad_w2     <<<(128 * 16 + TB - 1) / TB, TB, 0, stream>>>(W2, Wp);
    zero_f32   <<<(N * 2 + TB - 1) / TB, TB, 0, stream>>>(agg2, (size_t)N * 2);
    gemm_wmma  <<<gGemm, 256, 0, stream>>>(h, Wp, t2, N, 16, 16, 16);
    scatter2   <<<(E + 63) / 64, 128, 0, stream>>>(t2, src, dst, norm, agg2, E);
    finalize2  <<<(N * 2 + TB - 1) / TB, TB, 0, stream>>>(agg2, t2, sn, b2, out, N);
}
